// CoreDecoder_7911329759358
// MI455X (gfx1250) — matrix operand order repaired and checked by reading, not compiled, not run on hardware
//
#include <hip/hip_runtime.h>
#include <hip/hip_bf16.h>
#include <stdint.h>

typedef __attribute__((ext_vector_type(16))) __bf16 v16bf;
typedef __attribute__((ext_vector_type(8)))  float  v8f;
typedef __attribute__((ext_vector_type(4)))  int    v4i;

#define BT   51200   // B*T = 128*400
#define TT   400
#define XCH  736

union FragU { v16bf v; v4i q[2]; };

// ---------------- small device helpers ----------------

__device__ __forceinline__ unsigned short f2bf(float f) {
  unsigned u = __float_as_uint(f);
  u += 0x7FFFu + ((u >> 16) & 1u);          // round-to-nearest-even
  return (unsigned short)(u >> 16);
}

__device__ __forceinline__ float sigm(float x) { return 1.0f / (1.0f + expf(-x)); }

// deterministic hash-noise standing in for the JAX threefry noiser
__device__ __forceinline__ float nzf(float x, unsigned id, unsigned idx) {
  unsigned s = idx * 0x9E3779B9u ^ (id * 0x85EBCA77u);
  s ^= s >> 16; s *= 0x7FEB352Du; s ^= s >> 15; s *= 0x846CA68Bu; s ^= s >> 16;
  float u01 = (float)(s >> 8) * (1.0f / 16777216.0f);
  float y = x + (u01 - 0.5f) * (1.0f / 127.0f);
  return fminf(1.0f, fmaxf(-1.0f, y));
}

__device__ __forceinline__ v8f vzero() {
  v8f z = {0.f, 0.f, 0.f, 0.f, 0.f, 0.f, 0.f, 0.f};
  return z;
}

// A fragment: 16x32 bf16 tile, row-major source [row][k], lda in elements.
// Lane l: M = l&15; lanes 0-15 hold K 0-7 & 16-23, lanes 16-31 hold K 8-15 & 24-31.
__device__ __forceinline__ v16bf fragA(const unsigned short* A, int lda,
                                       int row0, int k0, int lane) {
  int m = lane & 15, half = lane >> 4;
  const char* p = (const char*)(A + (size_t)(row0 + m) * lda + k0 + half * 8);
  FragU f;
  f.q[0] = *(const v4i*)p;
  f.q[1] = *(const v4i*)(p + 32);
  return f.v;
}

// B fragment: 32x16 bf16 tile from weights stored [N][K] row-major (ldw = K).
// Lane l: N = l&15; lanes 0-15 hold K 0-7 & 16-23, lanes 16-31 hold K 8-15 & 24-31.
__device__ __forceinline__ v16bf fragB(const unsigned short* W, int ldw,
                                       int n0, int k0, int lane) {
  int n = lane & 15, half = lane >> 4;
  const char* p = (const char*)(W + (size_t)(n0 + n) * ldw + k0 + half * 8);
  FragU f;
  f.q[0] = *(const v4i*)p;
  f.q[1] = *(const v4i*)(p + 32);
  return f.v;
}

__device__ __forceinline__ v8f wmma_bf16(v16bf a, v16bf b, v8f c) {
  return __builtin_amdgcn_wmma_f32_16x16x32_bf16(false, a, false, b,
                                                 (short)0, c, false, false);
}

// MT x NT register-blocked GEMM: B fragments are loaded once per k-step and
// reused across MT M-tiles, amortizing weight traffic (loads/wmma ~ 1.3 at
// MT=2, NT=6 vs 2.3 unblocked).
template <int NT, int MT>
__device__ __forceinline__ void gemm_tiles_mt(v8f (&acc)[MT][NT],
                                              const unsigned short* A, int lda, int row0,
                                              const unsigned short* W, int ldw, int n0,
                                              int K, int lane) {
  for (int k0 = 0; k0 < K; k0 += 32) {
    v16bf b[NT];
#pragma unroll
    for (int t = 0; t < NT; ++t) b[t] = fragB(W, ldw, n0 + t * 16, k0, lane);
    v16bf a[MT];
#pragma unroll
    for (int m = 0; m < MT; ++m) a[m] = fragA(A, lda, row0 + m * 16, k0, lane);
#pragma unroll
    for (int m = 0; m < MT; ++m)
#pragma unroll
      for (int t = 0; t < NT; ++t) acc[m][t] = wmma_bf16(a[m], b[t], acc[m][t]);
  }
}

// ---------------- conversion / init kernels ----------------

__global__ void k_cvt(const float* __restrict__ s, unsigned short* __restrict__ d, int n) {
  int i = blockIdx.x * blockDim.x + threadIdx.x;
  if (i < n) d[i] = f2bf(s[i]);
}

// convert with K padding: src [rows][ks] -> dst [rows][kd], zero pad
__global__ void k_cvt_pad(const float* __restrict__ s, unsigned short* __restrict__ d,
                          int rows, int ks, int kd) {
  long i = (long)blockIdx.x * blockDim.x + threadIdx.x;
  long n = (long)rows * kd;
  if (i >= n) return;
  int r = (int)(i / kd), j = (int)(i % kd);
  d[i] = (j < ks) ? f2bf(s[(size_t)r * ks + j]) : (unsigned short)0;
}

// conv weight (32, cin, 2) -> W0[o][c], W1[o][c] bf16
__global__ void k_cvt_conv(const float* __restrict__ w,
                           unsigned short* __restrict__ w0,
                           unsigned short* __restrict__ w1, int n) {
  int i = blockIdx.x * blockDim.x + threadIdx.x;
  if (i >= n) return;
  w0[i] = f2bf(w[2 * i + 0]);
  w1[i] = f2bf(w[2 * i + 1]);
}

__global__ void k_init_hidden(const float* __restrict__ st, const float* __restrict__ W,
                              const float* __restrict__ b, float* __restrict__ hid) {
  int i = blockIdx.x * blockDim.x + threadIdx.x;
  if (i >= 128 * 128) return;
  int bb = i / 128, o = i % 128;
  float acc = b[o];
  for (int k = 0; k < 24; ++k) acc += st[bb * 24 + k] * W[o * 24 + k];
  hid[i] = tanhf(acc);
}

__global__ void k_init_gs(const float* __restrict__ hid, const float* __restrict__ W,
                          const float* __restrict__ b, float* __restrict__ gs) {
  int i = blockIdx.x * blockDim.x + threadIdx.x;
  if (i >= 128 * 480) return;
  int bb = i / 480, o = i % 480;
  float acc = b[o];
  for (int k = 0; k < 128; ++k) acc += hid[bb * 128 + k] * W[o * 128 + k];
  gs[i] = tanhf(acc);
}

// ---------------- GEMM kernels (wave32: one wave = MT*16 rows x NT*16 cols) ----------------

// C[pos][n] = A @ W^T + bias  (used for gi and output projection)
template <int NT, int MT>
__global__ void __launch_bounds__(128)
k_gemm_bias_store(const unsigned short* __restrict__ A, int lda,
                  const unsigned short* __restrict__ W, int K,
                  const float* __restrict__ bias,
                  float* __restrict__ C, int ldc, int ngroups) {
  int lane = threadIdx.x & 31;
  int gw = blockIdx.x * (blockDim.x >> 5) + (threadIdx.x >> 5);
  int row0 = (gw / ngroups) * MT * 16;
  int n0 = (gw % ngroups) * NT * 16;
  v8f acc[MT][NT];
#pragma unroll
  for (int m = 0; m < MT; ++m)
#pragma unroll
    for (int t = 0; t < NT; ++t) acc[m][t] = vzero();
  gemm_tiles_mt<NT, MT>(acc, A, lda, row0, W, K, n0, K, lane);
  int mb = (lane >> 4) * 8, n = lane & 15;
#pragma unroll
  for (int m = 0; m < MT; ++m)
#pragma unroll
    for (int t = 0; t < NT; ++t) {
      int col = n0 + t * 16 + n;
      float bv = bias[col];
#pragma unroll
      for (int v = 0; v < 8; ++v) {
        int pos = row0 + m * 16 + mb + v;
        C[(size_t)pos * ldc + col] = acc[m][t][v] + bv;
      }
    }
}

// y = noise(tanh(A @ W^T + bias)) written to x (f32) and xbf (bf16) at column xoff
template <int NT, int MT>
__global__ void __launch_bounds__(128)
k_gemm_tanh_noise(const unsigned short* __restrict__ A, int lda,
                  const unsigned short* __restrict__ W, int K,
                  const float* __restrict__ bias,
                  float* __restrict__ x, unsigned short* __restrict__ xbf,
                  int xoff, unsigned nid) {
  int lane = threadIdx.x & 31;
  int gw = blockIdx.x * (blockDim.x >> 5) + (threadIdx.x >> 5);
  int row0 = gw * MT * 16;
  v8f acc[MT][NT];
#pragma unroll
  for (int m = 0; m < MT; ++m)
#pragma unroll
    for (int t = 0; t < NT; ++t) acc[m][t] = vzero();
  gemm_tiles_mt<NT, MT>(acc, A, lda, row0, W, K, 0, K, lane);
  int mb = (lane >> 4) * 8, n = lane & 15;
#pragma unroll
  for (int m = 0; m < MT; ++m)
#pragma unroll
    for (int t = 0; t < NT; ++t) {
      int col = t * 16 + n;
      float bv = bias[col];
#pragma unroll
      for (int v = 0; v < 8; ++v) {
        int pos = row0 + m * 16 + mb + v;
        float y = tanhf(acc[m][t][v] + bv);
        y = nzf(y, nid, (unsigned)(pos * XCH + xoff + col));
        size_t o = (size_t)pos * XCH + xoff + col;
        x[o] = y;
        xbf[o] = f2bf(y);
      }
    }
}

// u = noise(g) elementwise, dual f32/bf16 write
__global__ void k_glu_u(const float* __restrict__ g, float* __restrict__ u,
                        unsigned short* __restrict__ ubf, unsigned nid) {
  long i = (long)blockIdx.x * blockDim.x + threadIdx.x;
  if (i >= (long)BT * 96) return;
  float y = nzf(g[i], nid, (unsigned)i);
  u[i] = y;
  ubf[i] = f2bf(y);
}

// y = noise(u * sigmoid(u @ Wg^T)), written at column xoff of x/xbf
__global__ void __launch_bounds__(128)
k_glu_gemm(const unsigned short* __restrict__ Aubf,
           const unsigned short* __restrict__ W,
           const float* __restrict__ uf,
           float* __restrict__ x, unsigned short* __restrict__ xbf,
           int xoff, unsigned nid) {
  const int NT = 6, MT = 2;
  int lane = threadIdx.x & 31;
  int gw = blockIdx.x * (blockDim.x >> 5) + (threadIdx.x >> 5);
  int row0 = gw * MT * 16;
  v8f acc[MT][NT];
#pragma unroll
  for (int m = 0; m < MT; ++m)
#pragma unroll
    for (int t = 0; t < NT; ++t) acc[m][t] = vzero();
  gemm_tiles_mt<NT, MT>(acc, Aubf, 96, row0, W, 96, 0, 96, lane);
  int mb = (lane >> 4) * 8, n = lane & 15;
#pragma unroll
  for (int m = 0; m < MT; ++m)
#pragma unroll
    for (int t = 0; t < NT; ++t) {
      int col = t * 16 + n;
#pragma unroll
      for (int v = 0; v < 8; ++v) {
        int pos = row0 + m * 16 + mb + v;
        float s = sigm(acc[m][t][v]);
        float y = uf[(size_t)pos * 96 + col] * s;
        y = nzf(y, nid, (unsigned)(pos * XCH + xoff + col));
        size_t o = (size_t)pos * XCH + xoff + col;
        x[o] = y;
        xbf[o] = f2bf(y);
      }
    }
}

// causal conv (kernel 2) as two-tap GEMM: y = tanh(x[t-1]@W0^T + x[t]@W1^T + b)
__global__ void __launch_bounds__(128)
k_conv(const unsigned short* __restrict__ xbf_in, int cin,
       const unsigned short* __restrict__ W0, const unsigned short* __restrict__ W1,
       const float* __restrict__ bias,
       float* __restrict__ x, unsigned short* __restrict__ xbf,
       int xoff, const unsigned short* __restrict__ zrow, unsigned nid) {
  const int NT = 2, MT = 2;  // N = 32, 32 rows per wave
  int lane = threadIdx.x & 31;
  int gw = blockIdx.x * (blockDim.x >> 5) + (threadIdx.x >> 5);
  int row0 = gw * MT * 16;
  v8f acc[MT][NT];
#pragma unroll
  for (int m = 0; m < MT; ++m)
#pragma unroll
    for (int t = 0; t < NT; ++t) acc[m][t] = vzero();

  // tap 1: current timestep
  gemm_tiles_mt<NT, MT>(acc, xbf_in, XCH, row0, W1, cin, 0, cin, lane);

  // tap 0: previous timestep (zeros at t == 0)
  {
    int ml = lane & 15, half = lane >> 4;
    for (int k0 = 0; k0 < cin; k0 += 32) {
      v16bf b[NT];
#pragma unroll
      for (int t = 0; t < NT; ++t) b[t] = fragB(W0, cin, t * 16, k0, lane);
#pragma unroll
      for (int m = 0; m < MT; ++m) {
        int pos = row0 + m * 16 + ml;
        bool tz = (pos % TT) == 0;
        const char* p = tz ? (const char*)zrow
                           : (const char*)(xbf_in + (size_t)(pos - 1) * XCH + k0 + half * 8);
        FragU f;
        f.q[0] = *(const v4i*)p;
        f.q[1] = *(const v4i*)(p + 32);
#pragma unroll
        for (int t = 0; t < NT; ++t) acc[m][t] = wmma_bf16(f.v, b[t], acc[m][t]);
      }
    }
  }

  int mb = (lane >> 4) * 8, n = lane & 15;
#pragma unroll
  for (int m = 0; m < MT; ++m)
#pragma unroll
    for (int t = 0; t < NT; ++t) {
      int col = t * 16 + n;
      float bv = bias[col];
#pragma unroll
      for (int v = 0; v < 8; ++v) {
        int pos = row0 + m * 16 + mb + v;
        float y = tanhf(acc[m][t][v] + bv);
        y = nzf(y, nid, (unsigned)(pos * XCH + xoff + col));
        size_t o = (size_t)pos * XCH + xoff + col;
        x[o] = y;
        xbf[o] = f2bf(y);
      }
    }
}

// ---------------- GRU recurrence: 8 blocks x 16 batch rows, h in LDS ----------------
// 96 threads = 3 waves; wave w owns columns [w*32, w*32+32) across all 3 gates.
// All 18 whh B-fragments (2 col-tiles x 3 gates x 3 k-steps) are preloaded into
// registers once; the 400-step serial loop touches only LDS (h) and gi (prefetched).
__global__ void __launch_bounds__(96)
k_scan(const float* __restrict__ gs, int layer,
       const unsigned short* __restrict__ whh, const float* __restrict__ bhh,
       const float* __restrict__ gi, float* __restrict__ gout) {
  __shared__ unsigned short h_bf[16 * 96];
  __shared__ float h_f[16 * 96];
  int tid = threadIdx.x;
  int lane = tid & 31, w = tid >> 5;
  int bblk = blockIdx.x;  // 0..7

  // preload recurrent weights into registers (18 x v16bf = 144 VGPRs; only 3
  // waves per block so the fat allocation is free)
  v16bf bw[18];
#pragma unroll
  for (int c = 0; c < 2; ++c)
#pragma unroll
    for (int gg = 0; gg < 3; ++gg)
#pragma unroll
      for (int ks = 0; ks < 3; ++ks)
        bw[(c * 3 + gg) * 3 + ks] =
            fragB(whh, 96, gg * 96 + w * 32 + c * 16, ks * 32, lane);

  float br[2], bz[2], bn[2];
#pragma unroll
  for (int c = 0; c < 2; ++c) {
    int jj = w * 32 + c * 16 + (lane & 15);
    br[c] = bhh[jj];
    bz[c] = bhh[96 + jj];
    bn[c] = bhh[192 + jj];
  }

  for (int i = tid; i < 16 * 96; i += 96) {
    int bb = i / 96, j = i % 96;
    float v = gs[(size_t)(bblk * 16 + bb) * 480 + layer * 96 + j];
    h_f[i] = v;
    h_bf[i] = f2bf(v);
  }
  __syncthreads();

  for (int t = 0; t < TT; ++t) {
    v8f acc[2][3];
#pragma unroll
    for (int c = 0; c < 2; ++c)
#pragma unroll
      for (int gg = 0; gg < 3; ++gg) acc[c][gg] = vzero();

#pragma unroll
    for (int ks = 0; ks < 3; ++ks) {
      v16bf a = fragA(h_bf, 96, 0, ks * 32, lane);
#pragma unroll
      for (int c = 0; c < 2; ++c)
#pragma unroll
        for (int gg = 0; gg < 3; ++gg)
          acc[c][gg] = wmma_bf16(a, bw[(c * 3 + gg) * 3 + ks], acc[c][gg]);
    }
    __syncthreads();  // all h_bf reads complete before updates

    int mb = (lane >> 4) * 8, n = lane & 15;
#pragma unroll
    for (int c = 0; c < 2; ++c) {
      int jj = w * 32 + c * 16 + n;
#pragma unroll
      for (int v = 0; v < 8; ++v) {
        int bb = mb + v;
        int b = bblk * 16 + bb;
        size_t gbase = ((size_t)b * TT + t) * 288;
        float ir = gi[gbase + jj];
        float iz = gi[gbase + 96 + jj];
        float in = gi[gbase + 192 + jj];
        // pull next timestep's gi row toward the caches (global_prefetch_b8)
        __builtin_prefetch((const char*)(gi + gbase + 288 + jj), 0, 3);
        float r = sigm(ir + acc[c][0][v] + br[c]);
        float zg = sigm(iz + acc[c][1][v] + bz[c]);
        float ng = tanhf(in + r * (acc[c][2][v] + bn[c]));
        float hold = h_f[bb * 96 + jj];
        float hnew = (1.0f - zg) * ng + zg * hold;
        h_f[bb * 96 + jj] = hnew;
        h_bf[bb * 96 + jj] = f2bf(hnew);
        gout[((size_t)b * TT + t) * 96 + jj] = hnew;
      }
    }
    __syncthreads();
  }
}

// ---------------- host ----------------

extern "C" void kernel_launch(void* const* d_in, const int* in_sizes, int n_in,
                              void* d_out, int out_size, void* d_ws, size_t ws_size,
                              hipStream_t stream) {
  static const int GRU_IN_[5]  = {96, 224, 352, 480, 608};
  static const int CONV_IN_[5] = {192, 320, 448, 576, 704};

  const float* z    = (const float*)d_in[0];
  const float* st   = (const float*)d_in[1];
  const float* d1w  = (const float*)d_in[2];
  const float* d1b  = (const float*)d_in[3];
  const float* hiw  = (const float*)d_in[4];
  const float* hib  = (const float*)d_in[5];
  const float* giw  = (const float*)d_in[6];
  const float* gib  = (const float*)d_in[7];
  const float* outw = (const float*)d_in[8];
  const float* outb = (const float*)d_in[9];
  const float *wih[5], *whh[5], *bih[5], *bhh[5], *cw[5], *cb[5], *glw[5];
  for (int i = 0; i < 5; ++i) {
    wih[i] = (const float*)d_in[10 + 7 * i];
    whh[i] = (const float*)d_in[11 + 7 * i];
    bih[i] = (const float*)d_in[12 + 7 * i];
    bhh[i] = (const float*)d_in[13 + 7 * i];
    cw[i]  = (const float*)d_in[14 + 7 * i];
    cb[i]  = (const float*)d_in[15 + 7 * i];
    glw[i] = (const float*)d_in[16 + 7 * i];
  }

  char* p = (char*)d_ws;
  auto carve = [&](size_t n) { char* r = p; p += (n + 255) & ~(size_t)255; return r; };

  float*          x    = (float*)carve((size_t)BT * XCH * 4);
  unsigned short* xbf  = (unsigned short*)carve((size_t)BT * XCH * 2);
  float*          gibf = (float*)carve((size_t)BT * 288 * 4);
  float*          gbuf = (float*)carve((size_t)BT * 96 * 4);
  float*          ubuf = (float*)carve((size_t)BT * 96 * 4);
  unsigned short* ubf  = (unsigned short*)carve((size_t)BT * 96 * 2);
  unsigned short* zbf  = (unsigned short*)carve((size_t)BT * 96 * 2);
  unsigned short* d1wbf = (unsigned short*)carve(96 * 96 * 2);
  unsigned short *wihbf[5], *whhbf[5], *glubf[5], *c0bf[5], *c1bf[5];
  for (int i = 0; i < 5; ++i) {
    wihbf[i] = (unsigned short*)carve((size_t)288 * GRU_IN_[i] * 2);
    whhbf[i] = (unsigned short*)carve(288 * 96 * 2);
    glubf[i] = (unsigned short*)carve(96 * 96 * 2);
    c0bf[i]  = (unsigned short*)carve((size_t)32 * CONV_IN_[i] * 2);
    c1bf[i]  = (unsigned short*)carve((size_t)32 * CONV_IN_[i] * 2);
  }
  unsigned short* outbf = (unsigned short*)carve(80 * 736 * 2);
  float*          hid   = (float*)carve(128 * 128 * 4);
  float*          gs    = (float*)carve(128 * 480 * 4);
  unsigned short* zrow  = (unsigned short*)carve(256);

  hipMemsetAsync(zrow, 0, 256, stream);

  // weight / activation conversions to bf16
  auto blocks = [](long n) { return (int)((n + 255) / 256); };
  k_cvt_pad<<<blocks(96 * 96), 256, 0, stream>>>(d1w, d1wbf, 96, 80, 96);
  k_cvt_pad<<<blocks((long)BT * 96), 256, 0, stream>>>(z, zbf, BT, 80, 96);
  for (int i = 0; i < 5; ++i) {
    k_cvt<<<blocks(288L * GRU_IN_[i]), 256, 0, stream>>>(wih[i], wihbf[i], 288 * GRU_IN_[i]);
    k_cvt<<<blocks(288L * 96), 256, 0, stream>>>(whh[i], whhbf[i], 288 * 96);
    k_cvt<<<blocks(96L * 96), 256, 0, stream>>>(glw[i], glubf[i], 96 * 96);
    k_cvt_conv<<<blocks(32L * CONV_IN_[i]), 256, 0, stream>>>(cw[i], c0bf[i], c1bf[i],
                                                              32 * CONV_IN_[i]);
  }
  k_cvt<<<blocks(80L * 736), 256, 0, stream>>>(outw, outbf, 80 * 736);

  // initial hidden states
  k_init_hidden<<<64, 256, 0, stream>>>(st, hiw, hib, hid);
  k_init_gs<<<240, 256, 0, stream>>>(hid, giw, gib, gs);

  // dense1: (BT,96) = tanh(z @ W^T + b), noised -> x[:, 0:96]
  // 1600 M-groups (MT=2) x 1 N-group, 4 waves/block
  k_gemm_tanh_noise<6, 2><<<400, 128, 0, stream>>>(zbf, 96, d1wbf, 96, d1b, x, xbf, 0, 1u);

  for (int i = 0; i < 5; ++i) {
    int din = GRU_IN_[i], cin = CONV_IN_[i];
    // gi = x[:, :din] @ wih^T + bih   (1600 M-groups x 3 N-groups of 96)
    k_gemm_bias_store<6, 2><<<1200, 128, 0, stream>>>(xbf, XCH, wihbf[i], din, bih[i],
                                                      gibf, 288, 3);
    // sequential GRU scan, batch-split across 8 blocks
    k_scan<<<8, 96, 0, stream>>>(gs, i, whhbf[i], bhh[i], gibf, gbuf);
    // GLU branch
    k_glu_u<<<blocks((long)BT * 96), 256, 0, stream>>>(gbuf, ubuf, ubf,
                                                       (unsigned)(3 * i + 2));
    k_glu_gemm<<<400, 128, 0, stream>>>(ubf, glubf[i], ubuf, x, xbf, din,
                                        (unsigned)(3 * i + 3));
    // causal conv branch
    k_conv<<<400, 128, 0, stream>>>(xbf, cin, c0bf[i], c1bf[i], cb[i], x, xbf, cin,
                                    zrow, (unsigned)(3 * i + 4));
  }

  // output projection straight into d_out ((B,T,80) == reshaped (B,1600,20))
  k_gemm_bias_store<5, 2><<<400, 128, 0, stream>>>(xbf, XCH, outbf, 736, outb,
                                                   (float*)d_out, 80, 1);
}
